// GatedMultiScaleRetention_30915174596861
// MI455X (gfx1250) — matrix operand order repaired and checked by reading, not compiled, not run on hardware
//
#include <hip/hip_runtime.h>
#include <hip/hip_bf16.h>
#include <stdint.h>

typedef _Float16 f16;
typedef __attribute__((ext_vector_type(16))) _Float16 v16h;
typedef __attribute__((ext_vector_type(8)))  float    v8f;
typedef __attribute__((ext_vector_type(4)))  unsigned int u32x4;

#define H_  8
#define HW_ 64
#define B_  2
#define C_  2048
#define D_  512   // H*HW
#define RT_ 4096  // B*C

// ---- feature detection for CDNA5 async/TDM data movers ----
#if __has_builtin(__builtin_amdgcn_global_load_async_to_lds_b128) && \
    __has_builtin(__builtin_amdgcn_s_wait_asynccnt)
#define USE_ASYNC_LDS 1
#else
#define USE_ASYNC_LDS 0
#endif

#if __has_builtin(__builtin_amdgcn_tensor_load_to_lds) && \
    __has_builtin(__builtin_amdgcn_s_wait_tensorcnt)
#define USE_TDM 1
#else
#define USE_TDM 0
#endif

// exact pointee spelling per hipcc diagnostic: int __attribute__((vector_size(16)))
typedef int v4i __attribute__((vector_size(16)));
typedef __attribute__((address_space(1))) v4i as1_v4i;
typedef __attribute__((address_space(3))) v4i as3_v4i;
typedef __attribute__((address_space(3))) void as3_void;

__device__ __forceinline__ as1_v4i* to_glb4(const void* p) {
  return (as1_v4i*)(uintptr_t)p;
}
__device__ __forceinline__ as3_v4i* to_lds4(void* p) {
  return (as3_v4i*)(uintptr_t)p;
}
__device__ __forceinline__ unsigned int lds_off_of(void* p) {
  return (unsigned int)(uintptr_t)(as3_void*)(uintptr_t)p;
}

#if USE_TDM
// Issue one TDM descriptor copying `ndwords` contiguous dwords from global
// memory to LDS. D# layout per CDNA5 ISA ch.8 (group0: count/lds/global/type,
// group1: data_size + dim0/tile0). Uniform per wave; EXEC ignored by TDM.
__device__ __forceinline__ void tdm_load_contig(void* lds_dst, const void* gsrc,
                                                unsigned int ndwords) {
  typedef unsigned int uv4 __attribute__((ext_vector_type(4)));
  typedef int iv4 __attribute__((ext_vector_type(4)));
  typedef int iv8 __attribute__((ext_vector_type(8)));
  unsigned long long ga = (unsigned long long)(uintptr_t)gsrc;
  unsigned int lds_off = lds_off_of(lds_dst);
  uv4 g0;
  g0.x = 1u;                                    // count = 1 (valid descriptor)
  g0.y = lds_off;                               // lds_addr
  g0.z = (unsigned int)ga;                      // global_addr[31:0]
  g0.w = (unsigned int)(ga >> 32) | (2u << 30); // global_addr[56:32] | type=2
  iv8 g1;
  g1[0] = (int)(2u << 16);                      // data_size = 4 bytes
  g1[1] = (int)((ndwords & 0xffffu) << 16);     // tensor_dim0[15:0] @ bits 63:48
  g1[2] = (int)((ndwords >> 16) | (1u << 16));  // tensor_dim0[31:16], tensor_dim1=1
  g1[3] = (int)((ndwords & 0xffffu) << 16);     // tile_dim0 @ bits 127:112
  g1[4] = 1;                                    // tile_dim1 = 1
  g1[5] = (int)ndwords;                         // tensor_dim0_stride[31:0]
  g1[6] = 0;
  g1[7] = 0;
  iv4 z4 = {0, 0, 0, 0};
#if defined(__clang_major__) && (__clang_major__ >= 23)
  iv8 z8 = {0, 0, 0, 0, 0, 0, 0, 0};
  __builtin_amdgcn_tensor_load_to_lds(g0, g1, z4, z4, z8, 0);
#else
  __builtin_amdgcn_tensor_load_to_lds(g0, g1, z4, z4, 0);
#endif
}
#endif

union ABPack { v16h v; f16 h[16]; u32x4 q[2]; };

// A-matrix 16x32 f16, wave32 layout (ISA 7.12.2): lane = g*16 + m,
// halves[0..7] = K g*8+0..7, halves[8..15] = K 16+g*8+0..7, row M = lane&15.
__device__ __forceinline__ v16h load_tileA(const f16* base, int ldb) {
  int lane = threadIdx.x & 31;
  int row = lane & 15, g = lane >> 4;
  const f16* p = base + row * ldb + g * 8;
  ABPack r;
  r.q[0] = *(const u32x4*)(p);
  r.q[1] = *(const u32x4*)(p + 16);
  return r.v;
}

// B-matrix 32x16 f16: lane = g*16 + n, halves[0..7] = K g*8+0..7, halves[8..15] = K 16+g*8+0..7.
// base must point at element (k=0, n=0); storage is n-major rows of ldb halves.
__device__ __forceinline__ v16h load_tileB(const f16* base, int ldb) {
  int lane = threadIdx.x & 31;
  int n = lane & 15, g = lane >> 4;
  const f16* p = base + n * ldb + g * 8;
  ABPack r;
  r.q[0] = *(const u32x4*)(p);
  r.q[1] = *(const u32x4*)(p + 16);
  return r.v;
}

__device__ __forceinline__ v8f wmma32(v16h a, v16h b, v8f c) {
  return __builtin_amdgcn_wmma_f32_16x16x32_f16(false, a, false, b, (short)0, c,
                                                false, false);
}

__device__ __forceinline__ v8f vzero() {
  v8f z = {0.f, 0.f, 0.f, 0.f, 0.f, 0.f, 0.f, 0.f};
  return z;
}

// ---------------- Kernel 0: weight transpose + f32->f16 ----------------
__global__ void k_prep(const float* Wq, const float* Wk, const float* Wv,
                       const float* Wg, const float* Wo,
                       f16* WqT, f16* WkT, f16* WvT, f16* WgT, f16* WoT) {
  int tid = blockIdx.x * blockDim.x + threadIdx.x;
  int nth = gridDim.x * blockDim.x;
  for (int idx = tid; idx < 3 * H_ * HW_ * HW_; idx += nth) {
    int t = idx / (H_ * HW_ * HW_);
    int r = idx % (H_ * HW_ * HW_);
    int h = r / (HW_ * HW_);
    int ij = r % (HW_ * HW_);
    int i = ij / HW_, j = ij % HW_;
    const float* src = (t == 0) ? Wq : (t == 1) ? Wk : Wv;
    f16* dst = (t == 0) ? WqT : (t == 1) ? WkT : WvT;
    dst[h * HW_ * HW_ + j * HW_ + i] = (f16)src[h * HW_ * HW_ + i * HW_ + j];
  }
  for (int idx = tid; idx < D_ * D_; idx += nth) {
    int jk = idx / D_, hi = idx % D_;
    WgT[(size_t)jk * D_ + hi] = (f16)Wg[(size_t)hi * D_ + jk];
    WoT[(size_t)jk * D_ + hi] = (f16)Wo[(size_t)hi * D_ + jk];
  }
}

// ------------- Kernel 1: QKV projection + rotary split (WMMA) -------------
__global__ void __launch_bounds__(128) k_qkv(const float* x, const float* theta,
    const f16* WqT, const f16* WkT, const f16* WvT,
    f16* Qc, f16* Qs, f16* Kc, f16* Ks, f16* Vt) {
  int b = blockIdx.z, h = blockIdx.y;
  int bh = b * H_ + h;
  int c0 = blockIdx.x * 64;
  int tid = threadIdx.x, lane = tid & 31, w = tid >> 5, g = lane >> 4;

  __shared__ __attribute__((aligned(16))) f16 xs[64][64];
  for (int i = tid; i < 64 * 64; i += 128) {
    int r = i >> 6, col = i & 63;
    xs[r][col] = (f16)x[(size_t)(b * C_ + c0 + r) * D_ + h * HW_ + col];
  }
  __syncthreads();

  int r0 = w * 16;
  v16h A0 = load_tileA(&xs[r0][0], 64);
  v16h A1 = load_tileA(&xs[r0][0] + 32, 64);
  const f16* wq = WqT + h * HW_ * HW_;
  const f16* wk = WkT + h * HW_ * HW_;
  const f16* wv = WvT + h * HW_ * HW_;

#pragma unroll
  for (int n = 0; n < 4; ++n) {
    v8f aq = wmma32(A1, load_tileB(wq + n * 16 * 64 + 32, 64),
             wmma32(A0, load_tileB(wq + n * 16 * 64, 64), vzero()));
    v8f ak = wmma32(A1, load_tileB(wk + n * 16 * 64 + 32, 64),
             wmma32(A0, load_tileB(wk + n * 16 * 64, 64), vzero()));
    v8f av = wmma32(A1, load_tileB(wv + n * 16 * 64 + 32, 64),
             wmma32(A0, load_tileB(wv + n * 16 * 64, 64), vzero()));
    int j = n * 16 + (lane & 15);
    float th = theta[h * HW_ + j];
#pragma unroll
    for (int v = 0; v < 8; ++v) {
      int c = c0 + r0 + v + 8 * g;
      float sv, cv;
      __sincosf((float)c * th, &sv, &cv);
      size_t qoff = ((size_t)bh * C_ + c) * HW_ + j;
      Qc[qoff] = (f16)(aq[v] * cv);
      Qs[qoff] = (f16)(aq[v] * sv);
      Kc[qoff] = (f16)(ak[v] * cv);
      Ks[qoff] = (f16)(ak[v] * sv);
      Vt[((size_t)bh * HW_ + j) * C_ + c] = (f16)av[v];  // V transposed [i][c]
    }
  }
}

// -------- Kernel 2: causal retention (flash-style, WMMA) + L2 norm --------
__global__ void __launch_bounds__(128) k_ret(const f16* Qc, const f16* Qs,
    const f16* Kc, const f16* Ks, const f16* Vt,
    const float* gamma, const float* qk_scale,
    f16* OHre, f16* OHim) {
  int b = blockIdx.z, h = blockIdx.y;
  int bh = b * H_ + h;
  int qt = blockIdx.x;
  int tid = threadIdx.x, lane = tid & 31, w = tid >> 5, g = lane >> 4;
  int r0 = qt * 64 + w * 16;  // absolute first query row of this wave

  __shared__ __attribute__((aligned(16))) f16 kc_s[64][64];
  __shared__ __attribute__((aligned(16))) f16 ks_s[64][64];
  __shared__ __attribute__((aligned(16))) f16 vt_s[64][64];
  __shared__ __attribute__((aligned(16))) f16 sre_s[4][16][32];
  __shared__ __attribute__((aligned(16))) f16 sim_s[4][16][32];

  const f16* qcb = Qc + ((size_t)bh * C_ + r0) * HW_;
  const f16* qsb = Qs + ((size_t)bh * C_ + r0) * HW_;
  v16h aQc0 = load_tileA(qcb, 64), aQc1 = load_tileA(qcb + 32, 64);
  v16h aQs0 = load_tileA(qsb, 64), aQs1 = load_tileA(qsb + 32, 64);

  float lg = __logf(gamma[h]);
  float inv = 1.0f / qk_scale[0];

  v8f ore[4], oim[4];
#pragma unroll
  for (int n = 0; n < 4; ++n) { ore[n] = vzero(); oim[n] = vzero(); }

  for (int kt = 0; kt <= qt; ++kt) {
    int d0 = kt * 64;

#if USE_TDM
    // Kc/Ks tiles are fully contiguous 8KB blocks: one TDM descriptor each,
    // issued by wave 0 only (TDM is a per-wave DMA, EXEC ignored).
    if (w == 0) {
      tdm_load_contig(&kc_s[0][0], Kc + ((size_t)bh * C_ + d0) * HW_, 2048);
      tdm_load_contig(&ks_s[0][0], Ks + ((size_t)bh * C_ + d0) * HW_, 2048);
    }
#else
    for (int i = tid * 8; i < 64 * 64; i += 128 * 8) {
      int r = i >> 6, col = i & 63;
      *(u32x4*)&kc_s[r][col] = *(const u32x4*)(Kc + ((size_t)bh * C_ + d0 + r) * HW_ + col);
      *(u32x4*)&ks_s[r][col] = *(const u32x4*)(Ks + ((size_t)bh * C_ + d0 + r) * HW_ + col);
    }
#endif

#if USE_ASYNC_LDS
    // Vt tile (row-strided): per-lane async b128 copies direct to LDS.
    for (int i = tid * 8; i < 64 * 64; i += 128 * 8) {
      int r = i >> 6, col = i & 63;
      __builtin_amdgcn_global_load_async_to_lds_b128(
          to_glb4(Vt + ((size_t)bh * HW_ + r) * C_ + d0 + col),
          to_lds4(&vt_s[r][col]), 0, 0);
    }
    __builtin_amdgcn_s_wait_asynccnt(0);
#else
    for (int i = tid * 8; i < 64 * 64; i += 128 * 8) {
      int r = i >> 6, col = i & 63;
      *(u32x4*)&vt_s[r][col] = *(const u32x4*)(Vt + ((size_t)bh * HW_ + r) * C_ + d0 + col);
    }
#endif

#if USE_TDM
    if (w == 0) __builtin_amdgcn_s_wait_tensorcnt(0);
#endif
    if (kt < qt)  // warm L2/WGP$ for next K tile -> global_prefetch_b8
      __builtin_prefetch(Kc + ((size_t)bh * C_ + d0 + 64) * HW_, 0, 1);
    __syncthreads();

#pragma unroll
    for (int chunk = 0; chunk < 2; ++chunk) {
      int dd0 = chunk * 32;
      if (d0 + dd0 <= r0 + 15) {  // any causal work in this 32-key chunk
#pragma unroll
        for (int sub = 0; sub < 2; ++sub) {
          int ds = dd0 + sub * 16;
          v16h Bkc0 = load_tileB(&kc_s[ds][0], 64);
          v16h Bkc1 = load_tileB(&kc_s[ds][0] + 32, 64);
          v16h Bks0 = load_tileB(&ks_s[ds][0], 64);
          v16h Bks1 = load_tileB(&ks_s[ds][0] + 32, 64);
          // S_re = Qc*Kc^T + Qs*Ks^T ; S_im = Qs*Kc^T - Qc*Ks^T
          v8f are = wmma32(aQs1, Bks1, wmma32(aQs0, Bks0,
                    wmma32(aQc1, Bkc1, wmma32(aQc0, Bkc0, vzero()))));
          v8f t1 = wmma32(aQs1, Bkc1, wmma32(aQs0, Bkc0, vzero()));
          v8f t2 = wmma32(aQc1, Bks1, wmma32(aQc0, Bks0, vzero()));
          int dcol = d0 + ds + (lane & 15);
          int scol = sub * 16 + (lane & 15);
#pragma unroll
          for (int v = 0; v < 8; ++v) {
            int c = r0 + v + 8 * g;
            int delta = c - dcol;
            float wgt = (delta < 0) ? 0.f : __expf(lg * (float)delta) * inv;
            sre_s[w][v + 8 * g][scol] = (f16)(are[v] * wgt);
            sim_s[w][v + 8 * g][scol] = (f16)((t1[v] - t2[v]) * wgt);
          }
        }
        // restage decayed scores (C/D layout -> A layout) through LDS
        v16h aSre = load_tileA(&sre_s[w][0][0], 32);
        v16h aSim = load_tileA(&sim_s[w][0][0], 32);
#pragma unroll
        for (int n = 0; n < 4; ++n) {
          v16h Bv = load_tileB(&vt_s[n * 16][dd0], 64);
          ore[n] = wmma32(aSre, Bv, ore[n]);
          oim[n] = wmma32(aSim, Bv, oim[n]);
        }
      }
    }
    __syncthreads();
  }

  // per-head complex L2 norm over the 64 cols (wave32 shfl_xor reduction)
#pragma unroll
  for (int v = 0; v < 8; ++v) {
    float s = 0.f;
#pragma unroll
    for (int n = 0; n < 4; ++n)
      s += ore[n][v] * ore[n][v] + oim[n][v] * oim[n][v];
    s += __shfl_xor(s, 1, 32);
    s += __shfl_xor(s, 2, 32);
    s += __shfl_xor(s, 4, 32);
    s += __shfl_xor(s, 8, 32);
    float rn = rsqrtf(s);
    int c = r0 + v + 8 * g;
    size_t row = (size_t)b * C_ + c;
#pragma unroll
    for (int n = 0; n < 4; ++n) {
      int j = h * HW_ + n * 16 + (lane & 15);
      OHre[row * D_ + j] = (f16)(ore[n][v] * rn);
      OHim[row * D_ + j] = (f16)(oim[n][v] * rn);
    }
  }
}

// ------- Kernel 3: gating GEMM [4096x512]@[512x512] + complex g*sig(2g) -------
__global__ void __launch_bounds__(128) k_gate(const f16* OHre, const f16* OHim,
    const f16* WgT, f16* Ure, f16* Uim) {
  int tid = threadIdx.x, lane = tid & 31, w = tid >> 5, g = lane >> 4;
  int row0 = blockIdx.x * 64 + w * 16;
  int j0 = blockIdx.y * 64;
  v8f gre[4], gim[4];
#pragma unroll
  for (int n = 0; n < 4; ++n) { gre[n] = vzero(); gim[n] = vzero(); }

  for (int k0 = 0; k0 < D_; k0 += 32) {
    v16h aR = load_tileA(OHre + (size_t)row0 * D_ + k0, D_);
    v16h aI = load_tileA(OHim + (size_t)row0 * D_ + k0, D_);
#pragma unroll
    for (int n = 0; n < 4; ++n) {
      v16h Bg = load_tileB(WgT + (size_t)(j0 + n * 16) * D_ + k0, D_);
      gre[n] = wmma32(aR, Bg, gre[n]);
      gim[n] = wmma32(aI, Bg, gim[n]);
    }
  }
#pragma unroll
  for (int n = 0; n < 4; ++n) {
    int jk = j0 + n * 16 + (lane & 15);
#pragma unroll
    for (int v = 0; v < 8; ++v) {
      int row = row0 + v + 8 * g;
      float ohre = (float)OHre[(size_t)row * D_ + jk];
      float ohim = (float)OHim[(size_t)row * D_ + jk];
      float gr = gre[n][v], gi = gim[n][v];
      // w = g / (1 + exp(-2g)), complex
      float er = __expf(-2.f * gr);
      float sn, cs;
      __sincosf(2.f * gi, &sn, &cs);
      float dr = 1.f + er * cs;
      float di = -er * sn;
      float den = dr * dr + di * di;
      float wr = (gr * dr + gi * di) / den;
      float wi = (gi * dr - gr * di) / den;
      Ure[(size_t)row * D_ + jk] = (f16)(ohre * wr - ohim * wi);
      Uim[(size_t)row * D_ + jk] = (f16)(ohre * wi + ohim * wr);
    }
  }
}

// -------- Kernel 4: output GEMM -> complex64 (interleaved re,im floats) --------
__global__ void __launch_bounds__(128) k_out(const f16* Ure, const f16* Uim,
    const f16* WoT, float* out) {
  int tid = threadIdx.x, lane = tid & 31, w = tid >> 5, g = lane >> 4;
  int row0 = blockIdx.x * 64 + w * 16;
  int j0 = blockIdx.y * 64;
  v8f ore[4], oim[4];
#pragma unroll
  for (int n = 0; n < 4; ++n) { ore[n] = vzero(); oim[n] = vzero(); }

  for (int k0 = 0; k0 < D_; k0 += 32) {
    v16h aR = load_tileA(Ure + (size_t)row0 * D_ + k0, D_);
    v16h aI = load_tileA(Uim + (size_t)row0 * D_ + k0, D_);
#pragma unroll
    for (int n = 0; n < 4; ++n) {
      v16h Bo = load_tileB(WoT + (size_t)(j0 + n * 16) * D_ + k0, D_);
      ore[n] = wmma32(aR, Bo, ore[n]);
      oim[n] = wmma32(aI, Bo, oim[n]);
    }
  }
#pragma unroll
  for (int n = 0; n < 4; ++n) {
    int kcol = j0 + n * 16 + (lane & 15);
#pragma unroll
    for (int v = 0; v < 8; ++v) {
      int row = row0 + v + 8 * g;
      out[((size_t)row * D_ + kcol) * 2 + 0] = ore[n][v];
      out[((size_t)row * D_ + kcol) * 2 + 1] = oim[n][v];
    }
  }
}

// ------------------------------ launch ------------------------------
extern "C" void kernel_launch(void* const* d_in, const int* in_sizes, int n_in,
                              void* d_out, int out_size, void* d_ws, size_t ws_size,
                              hipStream_t stream) {
  const float* x        = (const float*)d_in[0];
  const float* Wq       = (const float*)d_in[1];
  const float* Wk       = (const float*)d_in[2];
  const float* Wv       = (const float*)d_in[3];
  const float* theta    = (const float*)d_in[4];
  const float* gamma    = (const float*)d_in[5];
  // d_in[6] target_norm is dead in the reference
  const float* qk_scale = (const float*)d_in[7];
  const float* Wo       = (const float*)d_in[8];
  const float* Wg       = (const float*)d_in[9];

  constexpr size_t NWT = (size_t)H_ * HW_ * HW_;     // 32768
  constexpr size_t N5  = (size_t)D_ * D_;            // 262144
  constexpr size_t NQ  = (size_t)B_ * H_ * C_ * HW_; // 2097152
  constexpr size_t NR  = (size_t)RT_ * D_;           // 2097152

  char* ws = (char*)d_ws;
  size_t o = 0;
  f16* WqT = (f16*)(ws + o); o += NWT * 2;
  f16* WkT = (f16*)(ws + o); o += NWT * 2;
  f16* WvT = (f16*)(ws + o); o += NWT * 2;
  f16* WgT = (f16*)(ws + o); o += N5 * 2;
  f16* WoT = (f16*)(ws + o); o += N5 * 2;
  f16* Qc  = (f16*)(ws + o); o += NQ * 2;
  f16* Qs  = (f16*)(ws + o); o += NQ * 2;
  f16* Kc  = (f16*)(ws + o); o += NQ * 2;
  f16* Ks  = (f16*)(ws + o); o += NQ * 2;
  f16* Vt  = (f16*)(ws + o); o += NQ * 2;
  f16* OHre = (f16*)(ws + o); o += NR * 2;
  f16* OHim = (f16*)(ws + o); o += NR * 2;
  f16* Ure  = (f16*)(ws + o); o += NR * 2;
  f16* Uim  = (f16*)(ws + o); o += NR * 2;

  k_prep<<<256, 256, 0, stream>>>(Wq, Wk, Wv, Wg, Wo, WqT, WkT, WvT, WgT, WoT);
  k_qkv<<<dim3(C_ / 64, H_, B_), 128, 0, stream>>>(x, theta, WqT, WkT, WvT,
                                                   Qc, Qs, Kc, Ks, Vt);
  k_ret<<<dim3(C_ / 64, H_, B_), 128, 0, stream>>>(Qc, Qs, Kc, Ks, Vt,
                                                   gamma, qk_scale, OHre, OHim);
  k_gate<<<dim3(RT_ / 64, D_ / 64), 128, 0, stream>>>(OHre, OHim, WgT, Ure, Uim);
  k_out<<<dim3(RT_ / 64, D_ / 64), 128, 0, stream>>>(Ure, Uim, WoT, (float*)d_out);
}
